// ProteinMPNN_11132555231786
// MI455X (gfx1250) — matrix operand order repaired and checked by reading, not compiled, not run on hardware
//
#include <hip/hip_runtime.h>
#include <math.h>

// Problem constants (ProteinMPNN encoder layer)
#define B_  4
#define N_  2048
#define K_  48
#define C_  128
#define C3_ 384

typedef __attribute__((ext_vector_type(16))) __bf16 v16bf;
typedef __attribute__((ext_vector_type(8)))  float  v8f;

// ---------------------------------------------------------------------------
// CDNA5 WMMA: D(16x16,f32) = A(16x32,bf16) x B(32x16,bf16) + C
// ---------------------------------------------------------------------------
__device__ __forceinline__ v8f wmma_bf16(v16bf a, v16bf b, v8f c) {
  return __builtin_amdgcn_wmma_f32_16x16x32_bf16(false, a, false, b, (short)0, c,
                                                 false, false);
}

__device__ __forceinline__ float gelu_f(float x) {
  // exact (erf) GELU, matches torch.nn.GELU default
  return 0.5f * x * (1.0f + erff(x * 0.70710678118654752440f));
}

// Swizzled index into a sequence of 16x32 bf16 WMMA-A tiles stored lane-major,
// so each lane fetches its v16bf operand with one contiguous 32B LDS read.
// 16-bit A layout (ISA 7.12.2): lane = half*16 + m; element e:
//   e in 0..7  -> k = half*8 + e ;  e in 8..15 -> k = 16 + half*8 + (e-8)
// inverted: half=(k>>3)&1, e=(k>=16)*8 + (k&7)
__device__ __forceinline__ int a_swz(int m, int kglobal) {
  int tile = kglobal >> 5;
  int kl   = kglobal & 31;
  int half = (kl >> 3) & 1;
  int e    = ((kl & 16) >> 1) + (kl & 7);
  return tile * 512 + (half * 16 + m) * 16 + e;
}

// Same, for an 8-aligned channel group (c8 % 8 == 0): the 8 elements
// c8..c8+7 are contiguous in LDS starting at this index (16B aligned).
__device__ __forceinline__ int a_swz8(int m, int c8) {
  int tile = c8 >> 5;
  int kl   = c8 & 31;
  int half = (kl >> 3) & 1;
  int e0   = (kl & 16) >> 1;
  return tile * 512 + (half * 16 + m) * 16 + e0;
}

union bf8pack {
  __bf16 h[8];
  uint4  v;
};

// Store one wave's 16x16 f32 accumulator (D layout: lane = (m>=8)*16 + n,
// vgpr r = m&7) as GELU'd bf16 into an A-swizzled LDS activation buffer.
__device__ __forceinline__ void store_act(__bf16* dst, const v8f& acc, float bias,
                                          int lane, int kg) {
  int mbase = (lane >> 4) * 8;
#pragma unroll
  for (int r = 0; r < 8; ++r)
    dst[a_swz(mbase + r, kg)] = (__bf16)gelu_f(acc[r] + bias);
}

// K-dim pipelined GEMM over NT swizzled A-tiles in LDS against per-wave
// B-tiles in VGPRs: overlaps the next ds_load_b128 pair with the current WMMA.
template <int NT>
__device__ __forceinline__ v8f gemm_lds(const __bf16* A, const v16bf* w, int lane) {
  v8f acc = {0.f, 0.f, 0.f, 0.f, 0.f, 0.f, 0.f, 0.f};
  v16bf cur = *(const v16bf*)&A[lane * 16];
#pragma unroll
  for (int t = 0; t < NT; ++t) {
    v16bf nxt = (t + 1 < NT) ? *(const v16bf*)&A[(t + 1) * 512 + lane * 16] : cur;
    acc = wmma_bf16(cur, w[t], acc);
    cur = nxt;
  }
  return acc;
}

// ---------------------------------------------------------------------------
// Weight pre-pack: fp32 [Kdim,Ndim] row-major -> bf16 WMMA-B tiles.
// Tile order: n_tile*(Kdim/32) + k_tile.  Within tile, lane-major:
// lane = ((k_local>>3)&1)*16 + n_local, element e = (k_local>=16)*8 + k_local%8.
// ---------------------------------------------------------------------------
__global__ __launch_bounds__(256) void pack_weight_kernel(const float* __restrict__ W,
                                                          __bf16* __restrict__ out,
                                                          int Kdim, int Ndim) {
  int total = Kdim * Ndim;
  int pid = blockIdx.x * 256 + threadIdx.x;
  if (pid >= total) return;
  int nK = Kdim >> 5;
  int tile = pid >> 9, r = pid & 511;
  int lane = r >> 4, e = r & 15;
  int n_tile = tile / nK, k_tile = tile - n_tile * nK;
  int n = n_tile * 16 + (lane & 15);
  int k = k_tile * 32 + ((e >> 3) << 4) + ((lane >> 4) << 3) + (e & 7);
  out[pid] = (__bf16)W[k * Ndim + n];
}

__global__ __launch_bounds__(256) void f32_to_bf16_kernel(const float* __restrict__ in,
                                                          __bf16* __restrict__ out,
                                                          int n) {
  int i = blockIdx.x * 256 + threadIdx.x;
  if (i < n) out[i] = (__bf16)in[i];
}

// ---------------------------------------------------------------------------
// Row LayerNorm over C=128:  x = xa + scaleb*xb ; y = LN(x)*g + b ; y *= rowmask
// One wave per row (4 channels/lane), shuffle reductions (wave32).
// ---------------------------------------------------------------------------
__global__ __launch_bounds__(256) void ln_kernel(const float* __restrict__ xa,
                                                 const float* __restrict__ xb,
                                                 float scaleb,
                                                 const float* __restrict__ g,
                                                 const float* __restrict__ beta,
                                                 const float* __restrict__ rowmask,
                                                 float* __restrict__ out_f32,
                                                 __bf16* __restrict__ out_bf16,
                                                 int nrows) {
  int row = blockIdx.x * 8 + (threadIdx.x >> 5);
  int lane = threadIdx.x & 31;
  if (row >= nrows) return;
  float v[4];
  float s = 0.f;
#pragma unroll
  for (int j = 0; j < 4; ++j) {
    int c = j * 32 + lane;
    float t = xa[row * C_ + c];
    if (xb) t += scaleb * xb[row * C_ + c];
    v[j] = t;
    s += t;
  }
#pragma unroll
  for (int d = 1; d < 32; d <<= 1) s += __shfl_xor(s, d, 32);
  float mu = s * (1.f / C_);
  float var = 0.f;
#pragma unroll
  for (int j = 0; j < 4; ++j) { float dd = v[j] - mu; var += dd * dd; }
#pragma unroll
  for (int d = 1; d < 32; d <<= 1) var += __shfl_xor(var, d, 32);
  float rs = rsqrtf(var * (1.f / C_) + 1e-5f);
  float rm = rowmask ? rowmask[row] : 1.f;
#pragma unroll
  for (int j = 0; j < 4; ++j) {
    int c = j * 32 + lane;
    float o = ((v[j] - mu) * rs * g[c] + beta[c]) * rm;
    out_f32[row * C_ + c] = o;
    if (out_bf16) out_bf16[row * C_ + c] = (__bf16)o;
  }
}

// Branch-free h_EV tile build: [16 rows x 384] = 3 segments of 16x128.
// Each segment = 256 chunks of 8 channels = exactly one chunk per thread.
// self/gather segments are straight 16B bf16 copies; edge converts f32->bf16.
__device__ __forceinline__ void build_hEV(__bf16* A0, const __bf16* selfrow,
                                          const float* __restrict__ hE,
                                          const __bf16* __restrict__ nodes_b,
                                          const int* __restrict__ Eidx, int bn,
                                          int b, int kt) {
  int m  = threadIdx.x >> 4;
  int c8 = (threadIdx.x & 15) * 8;
  int kk = kt * 16 + m;
  // segment 0: broadcast self row (bf16 copy)
  *(uint4*)&A0[a_swz8(m, c8)] = *(const uint4*)(selfrow + c8);
  // segment 1: edge features (f32 -> bf16)
  {
    const float* src = hE + (bn * K_ + kk) * C_ + c8;
    float4 f0 = *(const float4*)src;
    float4 f1 = *(const float4*)(src + 4);
    bf8pack u;
    u.h[0] = (__bf16)f0.x; u.h[1] = (__bf16)f0.y;
    u.h[2] = (__bf16)f0.z; u.h[3] = (__bf16)f0.w;
    u.h[4] = (__bf16)f1.x; u.h[5] = (__bf16)f1.y;
    u.h[6] = (__bf16)f1.z; u.h[7] = (__bf16)f1.w;
    *(uint4*)&A0[a_swz8(m, C_ + c8)] = u.v;
  }
  // segment 2: gathered neighbor rows (bf16 copy)
  {
    int j = Eidx[bn * K_ + kk];
    *(uint4*)&A0[a_swz8(m, 2 * C_ + c8)] =
        *(const uint4*)(nodes_b + (b * N_ + j) * C_ + c8);
  }
}

// ---------------------------------------------------------------------------
// Node message: per (b,n), dh[b,n,:] = sum_k mask * (W3^T gelu(W2^T gelu(W1^T hEV)))
// 8 waves, each owns 16 output columns; 3 k-tiles of 16 neighbors.
// ---------------------------------------------------------------------------
__global__ __launch_bounds__(256) void node_msg_kernel(
    const __bf16* __restrict__ hVb, const float* __restrict__ hE,
    const int* __restrict__ Eidx, const float* __restrict__ maskA,
    const __bf16* __restrict__ packW,  // W1 tiles @0, W2 @96*512, W3 @128*512
    const float* __restrict__ b1, const float* __restrict__ b2,
    const float* __restrict__ b3, float* __restrict__ dh) {
  __shared__ __align__(64) __bf16 A0[12 * 512];  // 16x384 h_EV, A-swizzled
  __shared__ __align__(64) __bf16 A1[4 * 512];   // 16x128
  __shared__ __align__(64) __bf16 A2[4 * 512];   // 16x128

  int bn = blockIdx.x;              // flattened b*N + n
  int b = bn >> 11;                 // N_ = 2048
  int wave = threadIdx.x >> 5, lane = threadIdx.x & 31;
  int ncol = wave * 16 + (lane & 15);
  int mbase = (lane >> 4) * 8;

  // Per-wave weight slices resident in VGPRs (12+4+4 tiles = 160 VGPRs)
  v16bf w1[12], w2[4], w3[4];
  const __bf16* pW1 = packW;
  const __bf16* pW2 = packW + 96 * 512;
  const __bf16* pW3 = packW + 128 * 512;
#pragma unroll
  for (int t = 0; t < 12; ++t)
    w1[t] = *(const v16bf*)(pW1 + (wave * 12 + t) * 512 + lane * 16);
#pragma unroll
  for (int t = 0; t < 4; ++t)
    w2[t] = *(const v16bf*)(pW2 + (wave * 4 + t) * 512 + lane * 16);
#pragma unroll
  for (int t = 0; t < 4; ++t)
    w3[t] = *(const v16bf*)(pW3 + (wave * 4 + t) * 512 + lane * 16);

  float bias1 = b1[ncol], bias2 = b2[ncol], bias3 = b3[ncol];
  float csum[8];
#pragma unroll
  for (int r = 0; r < 8; ++r) csum[r] = 0.f;

  const __bf16* selfrow = hVb + bn * C_;

  for (int kt = 0; kt < 3; ++kt) {
    build_hEV(A0, selfrow, hE, hVb, Eidx, bn, b, kt);
    __syncthreads();

    v8f acc = gemm_lds<12>(A0, w1, lane);
    store_act(A1, acc, bias1, lane, ncol);
    __syncthreads();

    acc = gemm_lds<4>(A1, w2, lane);
    store_act(A2, acc, bias2, lane, ncol);
    __syncthreads();

    acc = gemm_lds<4>(A2, w3, lane);
    const float* ma = maskA + bn * K_ + kt * 16 + mbase;
#pragma unroll
    for (int r = 0; r < 8; ++r) csum[r] += ma[r] * (acc[r] + bias3);
    __syncthreads();
  }

  // Column sum over the 16 rows: lane r-components cover m=mbase..mbase+7,
  // combine lane with lane^16 to get full sum over k within this wave's cols.
  float tot = 0.f;
#pragma unroll
  for (int r = 0; r < 8; ++r) tot += csum[r];
  tot += __shfl_xor(tot, 16, 32);
  if (lane < 16) dh[bn * C_ + ncol] = tot;
}

// ---------------------------------------------------------------------------
// Node FFN: y = x + W_out^T gelu(W_in^T x) + b  (pre-LN2), 16 rows per block.
// ---------------------------------------------------------------------------
__global__ __launch_bounds__(256) void ffn_kernel(
    const float* __restrict__ xln, const __bf16* __restrict__ xlnb,
    const __bf16* __restrict__ pin,   // ff_in tiles  (128 tiles, nK=4)
    const __bf16* __restrict__ pout,  // ff_out tiles (128 tiles, nK=16)
    const float* __restrict__ bin, const float* __restrict__ bout,
    float* __restrict__ y) {
  __shared__ __align__(64) __bf16 Ax[4 * 512];   // 16x128 input
  __shared__ __align__(64) __bf16 Ah[16 * 512];  // 16x512 hidden
  int r0 = blockIdx.x * 16;
  int wave = threadIdx.x >> 5, lane = threadIdx.x & 31;
  int mbase = (lane >> 4) * 8;

  {
    int m = threadIdx.x >> 4;
    int c8 = (threadIdx.x & 15) * 8;
    *(uint4*)&Ax[a_swz8(m, c8)] = *(const uint4*)(xlnb + (r0 + m) * C_ + c8);
  }
  __syncthreads();

#pragma unroll
  for (int j = 0; j < 4; ++j) {
    int ht = wave * 4 + j;  // hidden 16-col group, 0..31
    v16bf wi[4];
#pragma unroll
    for (int t = 0; t < 4; ++t)
      wi[t] = *(const v16bf*)(pin + (ht * 4 + t) * 512 + lane * 16);
    v8f acc = gemm_lds<4>(Ax, wi, lane);
    store_act(Ah, acc, bin[ht * 16 + (lane & 15)], lane, ht * 16 + (lane & 15));
  }
  __syncthreads();

  v16bf wo[16];
#pragma unroll
  for (int t = 0; t < 16; ++t)
    wo[t] = *(const v16bf*)(pout + (wave * 16 + t) * 512 + lane * 16);
  v8f acc = gemm_lds<16>(Ah, wo, lane);

  int ncol = wave * 16 + (lane & 15);
  float bias = bout[ncol];
#pragma unroll
  for (int r = 0; r < 8; ++r) {
    int m = mbase + r;
    y[(r0 + m) * C_ + ncol] = xln[(r0 + m) * C_ + ncol] + acc[r] + bias;
  }
}

// ---------------------------------------------------------------------------
// Edge update: per (b,n,ktile of 16 edges), h_En = LN3(h_E + MLP(h_EV2)).
// LN3 fused via LDS staging (reuses the A0 buffer).
// ---------------------------------------------------------------------------
__global__ __launch_bounds__(256) void edge_msg_kernel(
    const float* __restrict__ hE, const __bf16* __restrict__ hVnb,
    const int* __restrict__ Eidx,
    const __bf16* __restrict__ packW,  // W11 @0, W12 @96*512, W13 @128*512
    const float* __restrict__ b1, const float* __restrict__ b2,
    const float* __restrict__ b3, const float* __restrict__ lng,
    const float* __restrict__ lnb, float* __restrict__ hEn) {
  __shared__ __align__(64) __bf16 A0[12 * 512];
  __shared__ __align__(64) __bf16 A1[4 * 512];
  __shared__ __align__(64) __bf16 A2[4 * 512];
  float* Ofull = (float*)A0;  // 16x128 f32 (8KB) reuses A0 (12KB) after GEMM1

  int bid = blockIdx.x;
  int bn = bid / 3, kt = bid - bn * 3;
  int b = bn >> 11;
  int wave = threadIdx.x >> 5, lane = threadIdx.x & 31;
  int ncol = wave * 16 + (lane & 15);
  int mbase = (lane >> 4) * 8;

  v16bf w1[12], w2[4], w3[4];
  const __bf16* pW1 = packW;
  const __bf16* pW2 = packW + 96 * 512;
  const __bf16* pW3 = packW + 128 * 512;
#pragma unroll
  for (int t = 0; t < 12; ++t)
    w1[t] = *(const v16bf*)(pW1 + (wave * 12 + t) * 512 + lane * 16);
#pragma unroll
  for (int t = 0; t < 4; ++t)
    w2[t] = *(const v16bf*)(pW2 + (wave * 4 + t) * 512 + lane * 16);
#pragma unroll
  for (int t = 0; t < 4; ++t)
    w3[t] = *(const v16bf*)(pW3 + (wave * 4 + t) * 512 + lane * 16);
  float bias1 = b1[ncol], bias2 = b2[ncol], bias3 = b3[ncol];

  build_hEV(A0, hVnb + bn * C_, hE, hVnb, Eidx, bn, b, kt);
  __syncthreads();

  v8f acc = gemm_lds<12>(A0, w1, lane);
  store_act(A1, acc, bias1, lane, ncol);
  __syncthreads();

  acc = gemm_lds<4>(A1, w2, lane);
  store_act(A2, acc, bias2, lane, ncol);
  __syncthreads();

  acc = gemm_lds<4>(A2, w3, lane);
  // residual add into LDS f32 staging
#pragma unroll
  for (int r = 0; r < 8; ++r) {
    int m = mbase + r;
    int grow = bn * K_ + kt * 16 + m;
    Ofull[m * C_ + ncol] = hE[grow * C_ + ncol] + acc[r] + bias3;
  }
  __syncthreads();

  // Fused LN3: 16 threads per row (xor masks 1,2,4,8 stay within the group)
  int row = threadIdx.x >> 4;
  int sub = threadIdx.x & 15;
  float vr[8];
  float s = 0.f;
#pragma unroll
  for (int j = 0; j < 8; ++j) {
    vr[j] = Ofull[row * C_ + sub + j * 16];
    s += vr[j];
  }
#pragma unroll
  for (int d = 1; d < 16; d <<= 1) s += __shfl_xor(s, d, 32);
  float mu = s * (1.f / C_);
  float var = 0.f;
#pragma unroll
  for (int j = 0; j < 8; ++j) { float dd = vr[j] - mu; var += dd * dd; }
#pragma unroll
  for (int d = 1; d < 16; d <<= 1) var += __shfl_xor(var, d, 32);
  float rs = rsqrtf(var * (1.f / C_) + 1e-5f);
  int grow = bn * K_ + kt * 16 + row;
#pragma unroll
  for (int j = 0; j < 8; ++j) {
    int c = sub + j * 16;
    hEn[grow * C_ + c] = (vr[j] - mu) * rs * lng[c] + lnb[c];
  }
}

// ---------------------------------------------------------------------------
// Host launch
// ---------------------------------------------------------------------------
extern "C" void kernel_launch(void* const* d_in, const int* in_sizes, int n_in,
                              void* d_out, int out_size, void* d_ws, size_t ws_size,
                              hipStream_t stream) {
  (void)in_sizes; (void)n_in; (void)out_size; (void)ws_size;
  const float* hV    = (const float*)d_in[0];
  const float* hE    = (const float*)d_in[1];
  const int*   Eidx  = (const int*)d_in[2];
  const float* maskV = (const float*)d_in[3];
  const float* maskA = (const float*)d_in[4];
  const float* W1w  = (const float*)d_in[5];  const float* W1b  = (const float*)d_in[6];
  const float* W2w  = (const float*)d_in[7];  const float* W2b  = (const float*)d_in[8];
  const float* W3w  = (const float*)d_in[9];  const float* W3b  = (const float*)d_in[10];
  const float* W11w = (const float*)d_in[11]; const float* W11b = (const float*)d_in[12];
  const float* W12w = (const float*)d_in[13]; const float* W12b = (const float*)d_in[14];
  const float* W13w = (const float*)d_in[15]; const float* W13b = (const float*)d_in[16];
  const float* ln1g = (const float*)d_in[17]; const float* ln1b = (const float*)d_in[18];
  const float* ln2g = (const float*)d_in[19]; const float* ln2b = (const float*)d_in[20];
  const float* ln3g = (const float*)d_in[21]; const float* ln3b = (const float*)d_in[22];
  const float* ffiw = (const float*)d_in[23]; const float* ffib = (const float*)d_in[24];
  const float* ffow = (const float*)d_in[25]; const float* ffob = (const float*)d_in[26];

  // Workspace layout (bytes): packed bf16 weights | hV bf16 | dh f32 |
  // hln1 f32 | hln1 bf16 | hVn bf16   (~15.3 MB total)
  char* ws = (char*)d_ws;
  __bf16* packW = (__bf16*)ws;                       // 576 tiles * 512 = 589824 B
  __bf16* hVb   = (__bf16*)(ws + 589824);            // 2 MB
  float*  dh    = (float*)(ws + 2686976);            // 4 MB
  float*  hln1  = (float*)(ws + 6881280);            // 4 MB
  __bf16* hln1b = (__bf16*)(ws + 11075584);          // 2 MB
  __bf16* hVnb  = (__bf16*)(ws + 13172736);          // 2 MB

  float* outV = (float*)d_out;                 // [B,N,C]
  float* outE = outV + B_ * N_ * C_;           // [B,N,K,C]

  const int NROWS = B_ * N_;

  // 1) pack all weights to bf16 WMMA-B tile layout (L2-resident afterwards)
  auto npack = [](int kd, int nd) { return (kd * nd + 255) / 256; };
  pack_weight_kernel<<<npack(C3_, C_), 256, 0, stream>>>(W1w,  packW +   0 * 512, C3_, C_);
  pack_weight_kernel<<<npack(C_,  C_), 256, 0, stream>>>(W2w,  packW +  96 * 512, C_,  C_);
  pack_weight_kernel<<<npack(C_,  C_), 256, 0, stream>>>(W3w,  packW + 128 * 512, C_,  C_);
  pack_weight_kernel<<<npack(C3_, C_), 256, 0, stream>>>(W11w, packW + 160 * 512, C3_, C_);
  pack_weight_kernel<<<npack(C_,  C_), 256, 0, stream>>>(W12w, packW + 256 * 512, C_,  C_);
  pack_weight_kernel<<<npack(C_,  C_), 256, 0, stream>>>(W13w, packW + 288 * 512, C_,  C_);
  pack_weight_kernel<<<npack(C_, 4 * C_), 256, 0, stream>>>(ffiw, packW + 320 * 512, C_, 4 * C_);
  pack_weight_kernel<<<npack(4 * C_, C_), 256, 0, stream>>>(ffow, packW + 448 * 512, 4 * C_, C_);

  // 2) bf16 copy of h_V for A-construction / gather
  f32_to_bf16_kernel<<<(NROWS * C_ + 255) / 256, 256, 0, stream>>>(hV, hVb, NROWS * C_);

  // 3) node messages: dh = sum_k mask * MLP(h_EV)
  node_msg_kernel<<<NROWS, 256, 0, stream>>>(hVb, hE, Eidx, maskA, packW,
                                             W1b, W2b, W3b, dh);

  // 4) LN1(h_V + dh/30) -> hln1 (f32 + bf16)
  ln_kernel<<<NROWS / 8, 256, 0, stream>>>(hV, dh, 1.f / 30.f, ln1g, ln1b,
                                           nullptr, hln1, hln1b, NROWS);

  // 5) FFN with residual -> outV (pre-LN2)
  ffn_kernel<<<NROWS / 16, 256, 0, stream>>>(hln1, hln1b, packW + 320 * 512,
                                             packW + 448 * 512, ffib, ffob, outV);

  // 6) LN2 in-place on outV, apply mask_V, emit bf16 h_Vn
  ln_kernel<<<NROWS / 8, 256, 0, stream>>>(outV, nullptr, 0.f, ln2g, ln2b,
                                           maskV, outV, hVnb, NROWS);

  // 7) edge update with fused LN3 -> outE
  edge_msg_kernel<<<NROWS * 3, 256, 0, stream>>>(hE, hVnb, Eidx, packW + 160 * 512,
                                                 W11b, W12b, W13b, ln3g, ln3b, outE);
}
